// QMatmul_85701777424723
// MI455X (gfx1250) — matrix operand order repaired and checked by reading, not compiled, not run on hardware
//
#include <hip/hip_runtime.h>
#include <hip/hip_bf16.h>

// ---------------------------------------------------------------------------
// Fake-quant int8 batched GEMM for MI455X (gfx1250, wave32, WMMA).
//   q1 = fq(A, per row of K=128)   -> f16 workspace (int8 codes exact in f16)
//   q2 = fq(B, per row of N=2048)  -> f16 workspace, transposed to [n][k]
//   C  = q1 @ q2  (v_wmma_f32_16x16x32_f16, f32 accum)
//   out = fq(C, per row of N=2048) -> fused: 16 full rows live in acc VGPRs
//
// GEMM workgroup = 512 threads = 16 wave32, 8 N-tiles per wave (64 acc VGPRs)
// so ~4 waves/SIMD stay resident and B tiles can be batch-loaded ahead of the
// WMMAs (avoids the s_wait_loadcnt 0 -> wmma serialization seen at 16 tiles).
// All GEMM-phase addresses are (loop-invariant base + immediate offset).
// ---------------------------------------------------------------------------

typedef __attribute__((ext_vector_type(16))) _Float16 v16h;
typedef __attribute__((ext_vector_type(8)))  _Float16 v8h;
typedef __attribute__((ext_vector_type(8)))  float    v8f;

__device__ __forceinline__ float fq_val(float v, float rs, float sc) {
    // clip(round(v/scale), -128, 127) * scale ; rintf == round-half-even
    return fminf(fmaxf(rintf(v * rs), -128.0f), 127.0f) * sc;
}

// ---------------------------------------------------------------------------
// Kernel 1: fake-quant A rows (length 128), store f16.
// grid = BH*M = 65536 blocks of 128 threads.
// ---------------------------------------------------------------------------
__global__ __launch_bounds__(128) void quant_rows_a(const float* __restrict__ x,
                                                    _Float16* __restrict__ q) {
    const int row = blockIdx.x;
    const int tid = threadIdx.x;
    __shared__ float red[128];
    const float v = x[(size_t)row * 128 + tid];
    red[tid] = fabsf(v);
    __syncthreads();
#pragma unroll
    for (int d = 64; d > 0; d >>= 1) {
        if (tid < d) red[tid] = fmaxf(red[tid], red[tid + d]);
        __syncthreads();
    }
    const float am = fmaxf(red[0], 1e-8f);
    const float sc = am * (1.0f / 127.0f);
    const float rs = 127.0f / am;
    q[(size_t)row * 128 + tid] = (_Float16)fq_val(v, rs, sc);
}

// ---------------------------------------------------------------------------
// Kernel 2: per-(b,h,k) scale for B (abs-max over N=2048).
// grid = BH*K = 4096 blocks of 256 threads.
// ---------------------------------------------------------------------------
__global__ __launch_bounds__(256) void rowscale_b(const float* __restrict__ x,
                                                  float* __restrict__ s2) {
    const int row = blockIdx.x;
    const int tid = threadIdx.x;
    const float* p = x + (size_t)row * 2048;
    float m0 = 0.0f;
#pragma unroll
    for (int i = 0; i < 8; ++i) m0 = fmaxf(m0, fabsf(p[tid + 256 * i]));
#pragma unroll
    for (int d = 1; d < 32; d <<= 1) m0 = fmaxf(m0, __shfl_xor(m0, d, 32));
    __shared__ float wred[8];
    if ((tid & 31) == 0) wred[tid >> 5] = m0;
    __syncthreads();
    if (tid == 0) {
        float am = 0.0f;
#pragma unroll
        for (int w = 0; w < 8; ++w) am = fmaxf(am, wred[w]);
        s2[row] = fmaxf(am, 1e-8f) * (1.0f / 127.0f);
    }
}

// ---------------------------------------------------------------------------
// Kernel 3: fake-quant B and transpose to q2T[bh][n][k] (f16) via LDS,
// so GEMM B fragments are contiguous 16B loads.
// grid = (64 n-tiles of 32, 32 bh), 256 threads.
// ---------------------------------------------------------------------------
__global__ __launch_bounds__(256) void quant_b_t(const float* __restrict__ x,
                                                 const float* __restrict__ s2,
                                                 _Float16* __restrict__ qBT) {
    const int nt = blockIdx.x;   // 32-column tile
    const int bh = blockIdx.y;
    const int tid = threadIdx.x;
    __shared__ _Float16 tile[128][33];   // +1 pad to break bank conflicts
    const int n_in = tid & 31;
    const int k0   = tid >> 5;           // 0..7
    const float* xb = x + ((size_t)bh * 128 + k0) * 2048 + nt * 32 + n_in;
    const float* sb = s2 + bh * 128 + k0;
#pragma unroll
    for (int i = 0; i < 16; ++i) {
        const int k = k0 + i * 8;        // 0..127
        const float sc = sb[i * 8];
        const float rs = 1.0f / sc;
        const float v = xb[(size_t)i * 8 * 2048];
        tile[k][n_in] = (_Float16)fq_val(v, rs, sc);
    }
    __syncthreads();
    // 32 rows x 128 halves out, 8-half (16B) chunks, coalesced b128 stores
    _Float16* ob = qBT + (((size_t)bh * 2048) + nt * 32) * 128;
#pragma unroll
    for (int i = 0; i < 2; ++i) {
        const int c  = i * 256 + tid;    // 0..511 chunks
        const int n  = c >> 4;           // 0..31
        const int kc = (c & 15) * 8;     // 0..120
        alignas(16) _Float16 tmp[8];
#pragma unroll
        for (int e = 0; e < 8; ++e) tmp[e] = tile[kc + e][n];
        *(uint4*)(ob + (size_t)n * 128 + kc) = *(const uint4*)tmp;
    }
}

// ---------------------------------------------------------------------------
// Kernel 4: WMMA GEMM + fused output row fake-quant.
// grid = (128 M-blocks, 32 bh); 512 threads = 16 wave32.
// Workgroup owns a 16 x 2048 output slab entirely in acc VGPRs:
//   wave w -> 8 tiles of 16x16 covering columns [w*128, w*128+128).
// ---------------------------------------------------------------------------
__global__ __launch_bounds__(512) void qmm_gemm(const _Float16* __restrict__ qA,
                                                const _Float16* __restrict__ qBT,
                                                float* __restrict__ out) {
    constexpr int Mtot = 2048, Ktot = 128, Ntot = 2048;
    constexpr int TILES = 8;               // N-tiles per wave
    const int mblock = blockIdx.x;
    const int bh     = blockIdx.y;
    const int tid    = threadIdx.x;
    const int wave   = tid >> 5;           // 0..15
    const int lane   = tid & 31;
    const int lane16 = lane & 15;
    const bool hi    = lane >= 16;
    const int nBase  = wave * (TILES * 16);   // 128 columns per wave

    // A fragment (16-bit A 16x32 layout): lane<16 -> M=lane, K = {0..7,16..23};
    // lane>=16 -> M=lane-16, K = {8..15,24..31}.  Loop-invariant base:
    const _Float16* aBase =
        qA + ((size_t)(bh * Mtot + mblock * 16 + lane16)) * Ktot + (hi ? 8 : 0);
    // B fragment (16-bit B 32x16 layout): lane<16 -> N=lane, K={0..7,16..23};
    // lane>=16 -> N=lane-16, K={8..15,24..31}.  qBT is [n][k]; tile stride = 2048
    // halves (4 KiB), k-step stride = 32 halves -> all immediate offsets.
    const _Float16* bBase =
        qBT + ((size_t)(bh * Ntot + nBase + lane16)) * Ktot + (hi ? 8 : 0);

    v8f acc[TILES];
#pragma unroll
    for (int j = 0; j < TILES; ++j)
#pragma unroll
        for (int e = 0; e < 8; ++e) acc[j][e] = 0.0f;

#pragma unroll
    for (int s = 0; s < 4; ++s) {
        if (s < 3) __builtin_prefetch(aBase + s * 32 + 32, 0, 1);  // global_prefetch_b8
        union { v16h v; v8h h[2]; } a;
        a.h[0] = *(const v8h*)(aBase + s * 32);
        a.h[1] = *(const v8h*)(aBase + s * 32 + 16);
#pragma unroll
        for (int j = 0; j < TILES; ++j) {
            union { v16h v; v8h h[2]; } b;
            b.h[0] = *(const v8h*)(bBase + j * 2048 + s * 32);
            b.h[1] = *(const v8h*)(bBase + j * 2048 + s * 32 + 16);
            acc[j] = __builtin_amdgcn_wmma_f32_16x16x32_f16(
                false, a.v, false, b.v, (short)0, acc[j], false, false);
        }
    }

    // ---- fused per-row abs-max over the full N=2048 slab -------------------
    __shared__ int rowmaxI[16];
    if (tid < 16) rowmaxI[tid] = 0;
    __syncthreads();
#pragma unroll
    for (int r = 0; r < 8; ++r) {
        // C/D layout: VGPR r holds (M=r, N=lane) for lanes 0-15 and
        // (M=r+8, N=lane-16) for lanes 16-31.
        float m0 = 0.0f;
#pragma unroll
        for (int j = 0; j < TILES; ++j) m0 = fmaxf(m0, fabsf(acc[j][r]));
#pragma unroll
        for (int d = 1; d < 16; d <<= 1) m0 = fmaxf(m0, __shfl_xor(m0, d, 32));
        if (lane16 == 0)  // lanes 0 and 16: one atomic per (row, wave)
            atomicMax(&rowmaxI[r + (hi ? 8 : 0)], __float_as_int(m0));
    }
    __syncthreads();

    // ---- fused output fake-quant + single-pass store -----------------------
    // Base pointer already includes the lane's row-half (hi -> rows 8..15);
    // per-(r,j) displacement r*Ntot + j*16 is a compile-time immediate.
    float* oBase = out + ((size_t)(bh * Mtot + mblock * 16 + (hi ? 8 : 0))) * Ntot
                 + nBase + lane16;
#pragma unroll
    for (int r = 0; r < 8; ++r) {
        const int m  = r + (hi ? 8 : 0);
        const float am = fmaxf(__int_as_float(rowmaxI[m]), 1e-8f);
        const float sc = am * (1.0f / 127.0f);
        const float rs = 127.0f / am;
#pragma unroll
        for (int j = 0; j < TILES; ++j)
            oBase[r * Ntot + j * 16] = fq_val(acc[j][r], rs, sc);
    }
}

// ---------------------------------------------------------------------------
// Host-side launcher
// ---------------------------------------------------------------------------
extern "C" void kernel_launch(void* const* d_in, const int* in_sizes, int n_in,
                              void* d_out, int out_size, void* d_ws, size_t ws_size,
                              hipStream_t stream) {
    (void)in_sizes; (void)n_in; (void)out_size; (void)ws_size;
    const float* in1 = (const float*)d_in[0];   // [2,16,2048,128]
    const float* in2 = (const float*)d_in[1];   // [2,16,128,2048]
    float* out = (float*)d_out;                 // [2,16,2048,2048]

    char* ws = (char*)d_ws;
    _Float16* qA  = (_Float16*)ws;                               // 16 MiB
    _Float16* qBT = (_Float16*)(ws + (size_t)16777216);          // 16 MiB
    float*    s2  = (float*)(ws + (size_t)2 * 16777216);         // 16 KiB

    quant_rows_a<<<dim3(65536),    dim3(128), 0, stream>>>(in1, qA);
    rowscale_b  <<<dim3(4096),     dim3(256), 0, stream>>>(in2, s2);
    quant_b_t   <<<dim3(64, 32),   dim3(256), 0, stream>>>(in2, s2, qBT);
    qmm_gemm    <<<dim3(128, 32),  dim3(512), 0, stream>>>(qA, qBT, out);
}
